// MultiHeadAttention_13434657702728
// MI455X (gfx1250) — compile-verified, matrix operand order checked by reading, not run on hardware
//
#include <hip/hip_runtime.h>
#include <hip/hip_bf16.h>
#include <cstdint>
#include <cstddef>

typedef __bf16 bf16_t;
typedef __attribute__((ext_vector_type(16))) __bf16 v16bf;
typedef __attribute__((ext_vector_type(8)))  float  v8f;

#define DIM   1024
#define HEADS 16
#define DH    64
#define BATCH 2
#define SEQ   2048
#define MROWS (BATCH*SEQ)   // 4096

union FragU { uint4 u[2]; v16bf v; };

// Load a 16-lane-striped bf16 WMMA fragment from a row-major [row][k] array.
// CDNA5 16-bit A/B layout: lane sub=lane&15 owns row (A) / column (B);
// half h=lane>>4 selects k-groups:  e=0..7 -> k0+8h+e ; e=8..15 -> k0+16+8h+(e-8).
// Both groups are 8 contiguous bf16 = one 16-byte load each.
__device__ __forceinline__ v16bf load_frag(const bf16_t* __restrict__ base, int row,
                                           int k0, int stride, int sub, int h) {
    const bf16_t* p = base + (size_t)(row + sub) * stride + (k0 + 8 * h);
    FragU f;
    f.u[0] = *(const uint4*)(p);
    f.u[1] = *(const uint4*)(p + 16);
    return f.v;
}

__device__ __forceinline__ v8f wmma_bf16(v16bf a, v16bf b, v8f c) {
    // (neg_a, A, neg_b, B, c_mod, C, reuse_a, reuse_b)
    return __builtin_amdgcn_wmma_f32_16x16x32_bf16(false, a, false, b, (short)0, c, false, false);
}

// ---------------- conversion kernels ----------------
__global__ void cvt_x_kernel(const float* __restrict__ x, bf16_t* __restrict__ xb, int n) {
    int i = blockIdx.x * blockDim.x + threadIdx.x;
    int stride = gridDim.x * blockDim.x;
    for (; i < n; i += stride) xb[i] = (bf16_t)x[i];
}

// Wt[n][k] = bf16(W[k][n])   (1024x1024) -> B operands become contiguous-in-k
__global__ void cvt_w_kernel(const float* __restrict__ W, bf16_t* __restrict__ Wt) {
    int i = blockIdx.x * blockDim.x + threadIdx.x;  // exactly 1M threads
    int n = i >> 10, k = i & 1023;
    Wt[i] = (bf16_t)W[k * DIM + n];
}

// ---------------- fused QKV projection ----------------
// Each wave computes a 32x16 tile of Q, K and V simultaneously:
// per k-step 2 A-frags + 3 B-frags feed 6 WMMAs (0.83 frag loads/WMMA).
// 6 accumulators (48 VGPRs) keeps the loop spill-free with good occupancy.
__global__ void qkv_proj_kernel(const bf16_t* __restrict__ xb,
                                const bf16_t* __restrict__ Wqt, const bf16_t* __restrict__ Wkt,
                                const bf16_t* __restrict__ Wvt,
                                const float* __restrict__ bq, const float* __restrict__ bk,
                                const float* __restrict__ bv,
                                bf16_t* __restrict__ Qb, bf16_t* __restrict__ Kb,
                                bf16_t* __restrict__ Vt) {
    int wave = threadIdx.x >> 5;
    int lane = threadIdx.x & 31;
    int sub = lane & 15, h = lane >> 4;
    int mblk  = blockIdx.x >> 3;                     // 128 blocks of 32 rows
    int ntile = ((blockIdx.x & 7) << 3) + wave;      // 64 tiles of 16 cols
    int m0 = mblk << 5, n0 = ntile << 4;

    v8f cq[2] = {}, ck[2] = {}, cv[2] = {};
    for (int k = 0; k < DIM; k += 32) {
        v16bf a0 = load_frag(xb, m0,      k, DIM, sub, h);
        v16bf a1 = load_frag(xb, m0 + 16, k, DIM, sub, h);
        v16bf bqf = load_frag(Wqt, n0, k, DIM, sub, h);
        v16bf bkf = load_frag(Wkt, n0, k, DIM, sub, h);
        v16bf bvf = load_frag(Wvt, n0, k, DIM, sub, h);
        cq[0] = wmma_bf16(a0, bqf, cq[0]);
        cq[1] = wmma_bf16(a1, bqf, cq[1]);
        ck[0] = wmma_bf16(a0, bkf, ck[0]);
        ck[1] = wmma_bf16(a1, bkf, ck[1]);
        cv[0] = wmma_bf16(a0, bvf, cv[0]);
        cv[1] = wmma_bf16(a1, bvf, cv[1]);
    }
    int n  = n0 + sub;
    int hh = n >> 6, d = n & 63;
    float biasq = bq[n], biask = bk[n], biasv = bv[n];
#pragma unroll
    for (int t = 0; t < 2; ++t) {
#pragma unroll
        for (int r = 0; r < 8; ++r) {
            int m = m0 + t * 16 + r + 8 * h;
            int b = m >> 11, s = m & 2047;
            size_t qk_idx = ((size_t)((b * HEADS + hh) * SEQ + s)) * DH + d;   // (b,h,s,d)
            Qb[qk_idx] = (bf16_t)(cq[t][r] + biasq);
            Kb[qk_idx] = (bf16_t)(ck[t][r] + biask);
            size_t v_idx = ((size_t)((b * HEADS + hh) * DH + d)) * SEQ + s;    // (b,h,d,s)
            Vt[v_idx] = (bf16_t)(cv[t][r] + biasv);
        }
    }
}

// ---------------- flash attention (online softmax) ----------------
// Scores computed TRANSPOSED (S^T = K_tile x Q^T) so that the probability
// tile converts in-register into the A fragment of P x V (no shuffles).
// Padding mask staged into LDS with CDNA5 async-to-LDS (ASYNCcnt) copies.
__global__ void attn_kernel(const bf16_t* __restrict__ Qb, const bf16_t* __restrict__ Kb,
                            const bf16_t* __restrict__ Vt, const int* __restrict__ mask,
                            bf16_t* __restrict__ AO) {
    __shared__ int smaski[SEQ];
    int qblk = blockIdx.x & 15;
    int h_   = (blockIdx.x >> 4) & 15;
    int b    = blockIdx.x >> 8;

    // Async copy of the 8KB mask row into LDS: 256 threads x 2 x b128.
    {
        int tid = threadIdx.x;
        uint64_t g0 = (uint64_t)(uintptr_t)(mask + b * SEQ + tid * 4);
        uint64_t g1 = (uint64_t)(uintptr_t)(mask + b * SEQ + tid * 4 + 1024);
        uint32_t l0 = (uint32_t)(uintptr_t)(&smaski[tid * 4]);
        uint32_t l1 = (uint32_t)(uintptr_t)(&smaski[tid * 4 + 1024]);
        asm volatile("global_load_async_to_lds_b128 %0, %1, off" :: "v"(l0), "v"(g0) : "memory");
        asm volatile("global_load_async_to_lds_b128 %0, %1, off" :: "v"(l1), "v"(g1) : "memory");
        asm volatile("s_wait_asynccnt 0" ::: "memory");
    }
    __syncthreads();

    int wave = threadIdx.x >> 5;
    int lane = threadIdx.x & 31;
    int sub = lane & 15, hl = lane >> 4;
    int q0 = qblk * 128 + wave * 16;

    const bf16_t* Qh = Qb + (size_t)(b * HEADS + h_) * SEQ * DH;
    const bf16_t* Kh = Kb + (size_t)(b * HEADS + h_) * SEQ * DH;
    const bf16_t* Vh = Vt + (size_t)(b * HEADS + h_) * DH * SEQ;

    // Q as B operand (B[k=d][n=q]): preload both d-halves, reused for all keys.
    v16bf qf0 = load_frag(Qh, q0, 0,  DH, sub, hl);
    v16bf qf1 = load_frag(Qh, q0, 32, DH, sub, hl);

    v8f O0 = {}, O1 = {}, O2 = {}, O3 = {};
    float mrow = -3.0e38f, lrow = 0.0f;
    const float scale = 0.125f;  // 1/sqrt(64)

    union MaskU { int4 v4[2]; int s[8]; };

    for (int j = 0; j < SEQ; j += 32) {
        // S^T tiles: D[key][q]; lane=q, VGPR r = key r+8*hl within 16-key tile
        v16bf ka0 = load_frag(Kh, j, 0,  DH, sub, hl);
        v16bf ka1 = load_frag(Kh, j, 32, DH, sub, hl);
        v8f t0 = {};
        t0 = wmma_bf16(ka0, qf0, t0);
        t0 = wmma_bf16(ka1, qf1, t0);
        v16bf kb0 = load_frag(Kh, j + 16, 0,  DH, sub, hl);
        v16bf kb1 = load_frag(Kh, j + 16, 32, DH, sub, hl);
        v8f t1 = {};
        t1 = wmma_bf16(kb0, qf0, t1);
        t1 = wmma_bf16(kb1, qf1, t1);

        // mask bits for this lane's 8+8 key rows: two ds_load_b128 pairs
        MaskU ma, mb;
        ma.v4[0] = *(const int4*)(smaski + j + 8 * hl);
        ma.v4[1] = *(const int4*)(smaski + j + 8 * hl + 4);
        mb.v4[0] = *(const int4*)(smaski + j + 16 + 8 * hl);
        mb.v4[1] = *(const int4*)(smaski + j + 16 + 8 * hl + 4);

        float mloc = -3.0e38f;
#pragma unroll
        for (int r = 0; r < 8; ++r) {
            float s0 = t0[r] * scale + (ma.s[r] ? -1.0e30f : 0.0f);
            float s1 = t1[r] * scale + (mb.s[r] ? -1.0e30f : 0.0f);
            t0[r] = s0; t1[r] = s1;
            mloc = fmaxf(mloc, fmaxf(s0, s1));
        }
        // per-query stats: keys 0-7 of q live in lane q, keys 8-15 in lane q+16
        mloc = fmaxf(mloc, __shfl_xor(mloc, 16));
        float mnew  = fmaxf(mrow, mloc);
        float alpha = __expf(mrow - mnew);
        float psum  = 0.0f;
        v16bf pfrag = {};
#pragma unroll
        for (int r = 0; r < 8; ++r) {
            float p0 = __expf(t0[r] - mnew);
            float p1 = __expf(t1[r] - mnew);
            psum += p0 + p1;
            pfrag[r]     = (bf16_t)p0;  // k = 8*hl + r        (keys j..j+15 tile)
            pfrag[r + 8] = (bf16_t)p1;  // k = 16 + 8*hl + r   (keys j+16.. tile)
        }
        psum += __shfl_xor(psum, 16);
        lrow = lrow * alpha + psum;
        mrow = mnew;

        // rescale O: row m = r + 8*hl needs alpha of that query (lives at lane r+8*hl)
#pragma unroll
        for (int r = 0; r < 8; ++r) {
            float arow = __shfl(alpha, r + 8 * hl);
            O0[r] *= arow; O1[r] *= arow; O2[r] *= arow; O3[r] *= arow;
        }
        // P x V: B[k=key][n=d] from V^T (b,h,d,s)
        v16bf vf0 = load_frag(Vh, 0,  j, SEQ, sub, hl);
        v16bf vf1 = load_frag(Vh, 16, j, SEQ, sub, hl);
        v16bf vf2 = load_frag(Vh, 32, j, SEQ, sub, hl);
        v16bf vf3 = load_frag(Vh, 48, j, SEQ, sub, hl);
        O0 = wmma_bf16(pfrag, vf0, O0);
        O1 = wmma_bf16(pfrag, vf1, O1);
        O2 = wmma_bf16(pfrag, vf2, O2);
        O3 = wmma_bf16(pfrag, vf3, O3);
    }

    float linv = 1.0f / lrow;
#pragma unroll
    for (int r = 0; r < 8; ++r) {
        float li = __shfl(linv, r + 8 * hl);
        int q = q0 + r + 8 * hl;
        size_t rowoff = ((size_t)(b * SEQ + q)) * DIM + h_ * DH;
        AO[rowoff + 0 * 16 + sub] = (bf16_t)(O0[r] * li);
        AO[rowoff + 1 * 16 + sub] = (bf16_t)(O1[r] * li);
        AO[rowoff + 2 * 16 + sub] = (bf16_t)(O2[r] * li);
        AO[rowoff + 3 * 16 + sub] = (bf16_t)(O3[r] * li);
    }
}

// ---------------- output projection ----------------
// Each wave computes a 32x32 tile: per k-step 2 A-frags + 2 B-frags -> 4 WMMAs.
__global__ void out_proj_kernel(const bf16_t* __restrict__ AO, const bf16_t* __restrict__ Wot,
                                const float* __restrict__ bo, float* __restrict__ out) {
    int wave = threadIdx.x >> 5;
    int lane = threadIdx.x & 31;
    int sub = lane & 15, h = lane >> 4;
    int mblk = blockIdx.x >> 2;                    // 128 blocks of 32 rows
    int nblk = ((blockIdx.x & 3) << 3) + wave;     // 32 blocks of 32 cols
    int m0 = mblk << 5, n0 = nblk << 5;

    v8f c0[2] = {}, c1[2] = {};
    for (int k = 0; k < DIM; k += 32) {
        v16bf a0 = load_frag(AO, m0,      k, DIM, sub, h);
        v16bf a1 = load_frag(AO, m0 + 16, k, DIM, sub, h);
        v16bf b0 = load_frag(Wot, n0,      k, DIM, sub, h);
        v16bf b1 = load_frag(Wot, n0 + 16, k, DIM, sub, h);
        c0[0] = wmma_bf16(a0, b0, c0[0]);
        c0[1] = wmma_bf16(a1, b0, c0[1]);
        c1[0] = wmma_bf16(a0, b1, c1[0]);
        c1[1] = wmma_bf16(a1, b1, c1[1]);
    }
    float biasA = bo[n0 + sub];
    float biasB = bo[n0 + 16 + sub];
#pragma unroll
    for (int t = 0; t < 2; ++t) {
#pragma unroll
        for (int r = 0; r < 8; ++r) {
            int m = m0 + t * 16 + r + 8 * h;
            out[(size_t)m * DIM + n0 + sub]      = c0[t][r] + biasA;
            out[(size_t)m * DIM + n0 + 16 + sub] = c1[t][r] + biasB;
        }
    }
}

extern "C" void kernel_launch(void* const* d_in, const int* in_sizes, int n_in,
                              void* d_out, int out_size, void* d_ws, size_t ws_size,
                              hipStream_t stream) {
    (void)in_sizes; (void)n_in; (void)out_size; (void)ws_size;
    const float* x    = (const float*)d_in[0];
    const int*   mask = (const int*)d_in[1];
    const float* Wq   = (const float*)d_in[2];
    const float* bq   = (const float*)d_in[3];
    const float* Wk   = (const float*)d_in[4];
    const float* bk   = (const float*)d_in[5];
    const float* Wv   = (const float*)d_in[6];
    const float* bv   = (const float*)d_in[7];
    const float* Wo   = (const float*)d_in[8];
    const float* bo   = (const float*)d_in[9];
    float* out = (float*)d_out;

    char* ws = (char*)d_ws;                       // 48 MB used
    bf16_t* xb  = (bf16_t*)(ws);                  //  8 MB  x  bf16 (4096x1024)
    bf16_t* Wqt = (bf16_t*)(ws + (8u  << 20));    //  2 MB  Wq^T bf16
    bf16_t* Wkt = (bf16_t*)(ws + (10u << 20));    //  2 MB
    bf16_t* Wvt = (bf16_t*)(ws + (12u << 20));    //  2 MB
    bf16_t* Wot = (bf16_t*)(ws + (14u << 20));    //  2 MB
    bf16_t* Qb  = (bf16_t*)(ws + (16u << 20));    //  8 MB  (b,h,s,d)
    bf16_t* Kb  = (bf16_t*)(ws + (24u << 20));    //  8 MB  (b,h,s,d)
    bf16_t* Vt  = (bf16_t*)(ws + (32u << 20));    //  8 MB  (b,h,d,s)
    bf16_t* AO  = (bf16_t*)(ws + (40u << 20));    //  8 MB  (b,s,h*d)

    cvt_x_kernel<<<2048, 256, 0, stream>>>(x, xb, MROWS * DIM);
    cvt_w_kernel<<<4096, 256, 0, stream>>>(Wq, Wqt);
    cvt_w_kernel<<<4096, 256, 0, stream>>>(Wk, Wkt);
    cvt_w_kernel<<<4096, 256, 0, stream>>>(Wv, Wvt);
    cvt_w_kernel<<<4096, 256, 0, stream>>>(Wo, Wot);
    qkv_proj_kernel<<<1024, 256, 0, stream>>>(xb, Wqt, Wkt, Wvt, bq, bk, bv, Qb, Kb, Vt);
    attn_kernel<<<512, 256, 0, stream>>>(Qb, Kb, Vt, mask, AO);
    out_proj_kernel<<<512, 256, 0, stream>>>(AO, Wot, bo, out);
}
